// SelfAttention_36893769072658
// MI455X (gfx1250) — matrix operand order repaired and checked by reading, not compiled, not run on hardware
//
#include <hip/hip_runtime.h>

// ---------------------------------------------------------------------------
// CDNA5 (gfx1250) self-attention block, wave32 + v_wmma_f32_16x16x32_bf16.
// ---------------------------------------------------------------------------

typedef __attribute__((ext_vector_type(16))) __bf16 v16bf;
typedef __attribute__((ext_vector_type(8)))  float  v8f;

union Frag32B { uint4 q[2]; v16bf h; };

#define WMMA_BF16(A, B, C) \
    __builtin_amdgcn_wmma_f32_16x16x32_bf16(false, (A), false, (B), (short)0, (C), false, false)

// LDS column swizzle for the 16x1024 score tile: XOR in 8-float blocks keeps
// float4 runs contiguous/aligned while spreading row starts across banks.
#define SW(row, col) ((row) * 1024 + ((col) ^ ((row) << 3)))

// A-matrix fragment (16x32 bf16). Memory: row-major, K contiguous, row stride ldk.
// lane l: m = l&15, kg = (l>=16)?16:0 ; elems 0..15 -> k=[kg,kg+16)
__device__ __forceinline__ v16bf load_a_frag(const __bf16* base, int ldk, int m, int kg) {
    Frag32B f;
    f.q[0] = *(const uint4*)(base + (size_t)m * ldk + kg);
    f.q[1] = *(const uint4*)(base + (size_t)m * ldk + kg + 8);
    return f.h;
}

// B-matrix fragment (32x16 bf16). Memory: column(n)-major rows, K contiguous, row stride ldk.
// lane l: n = l&15, kb = (l>=16)?16:0 ; elems 0..15 -> k=[kb,kb+16)
__device__ __forceinline__ v16bf load_b_frag(const __bf16* base, int ldk, int n, int kb) {
    Frag32B f;
    const uint4* p = (const uint4*)(base + (size_t)n * ldk + kb);
    f.q[0] = p[0];
    f.q[1] = p[1];
    return f.h;
}

// pack 8 f32 -> 8 bf16 (16 bytes)
__device__ __forceinline__ uint4 pack8_bf16(const float* f) {
    union { __bf16 h[8]; uint4 u; } r;
#pragma unroll
    for (int i = 0; i < 8; i++) r.h[i] = (__bf16)f[i];
    return r.u;
}

// ---------------------------------------------------------------------------
// Kernel 0: f32 -> bf16 weight conversion
// ---------------------------------------------------------------------------
__global__ void cvt_bf16_kernel(const float* __restrict__ w, __bf16* __restrict__ o, int n) {
    int i = blockIdx.x * 256 + threadIdx.x;
    if (i < n) o[i] = (__bf16)w[i];
}

// ---------------------------------------------------------------------------
// Kernel 1: GroupNorm (8 groups) + affine, output transposed (b, s, c) bf16.
// grid = 64 blocks of 256. Transpose goes through a padded LDS tile so global
// writes are 128B-contiguous per spatial row.
// ---------------------------------------------------------------------------
__global__ __launch_bounds__(256) void gn_kernel(const float* __restrict__ x,
                                                 const float* __restrict__ gamma,
                                                 const float* __restrict__ beta,
                                                 __bf16* __restrict__ xnt) {
    const int b = blockIdx.x >> 3;
    const int g = blockIdx.x & 7;
    const float* xp = x + ((size_t)(b * 512 + g * 64)) * 1024;

    float s = 0.f, ss = 0.f;
    for (int i = threadIdx.x; i < 65536; i += 256) {
        float v = xp[i];
        s += v; ss += v * v;
    }
    __shared__ float rs[256], rss[256];
    __shared__ float tile[64 * 65];               // pad 65: conflict-free transpose
    rs[threadIdx.x] = s; rss[threadIdx.x] = ss;
    __syncthreads();
    for (int o = 128; o > 0; o >>= 1) {
        if (threadIdx.x < o) {
            rs[threadIdx.x]  += rs[threadIdx.x + o];
            rss[threadIdx.x] += rss[threadIdx.x + o];
        }
        __syncthreads();
    }
    const float mean = rs[0] * (1.0f / 65536.0f);
    const float var  = rss[0] * (1.0f / 65536.0f) - mean * mean;
    const float inv  = rsqrtf(var + 1e-5f);

    for (int sblk = 0; sblk < 16; sblk++) {
        for (int i = threadIdx.x; i < 4096; i += 256) {     // coalesced read over s
            int c_l = i >> 6, s_l = i & 63;
            tile[c_l * 65 + s_l] = xp[(size_t)c_l * 1024 + sblk * 64 + s_l];
        }
        __syncthreads();
        for (int i = threadIdx.x; i < 4096; i += 256) {     // coalesced write over c
            int s_l = i >> 6, c_l = i & 63;
            int c   = g * 64 + c_l;
            float v = (tile[c_l * 65 + s_l] - mean) * inv * gamma[c] + beta[c];
            xnt[((size_t)b * 1024 + sblk * 64 + s_l) * 512 + c] = (__bf16)v;
        }
        __syncthreads();
    }
}

// ---------------------------------------------------------------------------
// Kernel 2: QKV GEMM  qkv[o,s] = sum_c W[o,c] * xnt[b,s,c] + bias[o]
// one wave per 64(o) x 64(s) tile: 4 A-frags x 4 B-frags -> 16 WMMA / K-step.
// grid = 8 * 24 * 16 = 3072 blocks of 32.
// Epilogue transposes through padded LDS for wide contiguous bf16 stores.
// outputs: q,k as (b,h,s,d) bf16 ; v as (b,h,d,t) bf16
// ---------------------------------------------------------------------------
__global__ __launch_bounds__(32) void qkv_gemm_kernel(const __bf16* __restrict__ wq,
                                                      const __bf16* __restrict__ xnt,
                                                      const float*  __restrict__ bias,
                                                      __bf16* __restrict__ qb,
                                                      __bf16* __restrict__ kb,
                                                      __bf16* __restrict__ vb) {
    __shared__ float T[64 * 68];                  // 17.4 KB transpose tile

    const int l = threadIdx.x;
    const int m = l & 15, half = l >> 4;
    const int kg = half * 16, kb16 = half * 16;
    int id = blockIdx.x;
    const int st4 = id & 15; id >>= 4;
    const int ot = id % 24;
    const int b  = id / 24;
    const int o0 = ot * 64, s0 = st4 * 64;

    v8f acc[4][4] = {};
    const __bf16* wrow = wq  + (size_t)o0 * 512;
    const __bf16* xb   = xnt + ((size_t)b * 1024 + s0) * 512;

    for (int kc = 0; kc < 512; kc += 32) {
        __builtin_prefetch(wrow + (size_t)(m * 4) * 512 + kc + 64, 0, 1);
        __builtin_prefetch(xb + (size_t)(m * 4) * 512 + kc + 64, 0, 1);
        v16bf a0 = load_a_frag(wrow + kc,                     512, m, kg);
        v16bf a1 = load_a_frag(wrow + (size_t)16 * 512 + kc,  512, m, kg);
        v16bf a2 = load_a_frag(wrow + (size_t)32 * 512 + kc,  512, m, kg);
        v16bf a3 = load_a_frag(wrow + (size_t)48 * 512 + kc,  512, m, kg);
#pragma unroll
        for (int j = 0; j < 4; j++) {
            v16bf bf = load_b_frag(xb + (size_t)j * 16 * 512 + kc, 512, m, kb16);
            acc[0][j] = WMMA_BF16(a0, bf, acc[0][j]);
            acc[1][j] = WMMA_BF16(a1, bf, acc[1][j]);
            acc[2][j] = WMMA_BF16(a2, bf, acc[2][j]);
            acc[3][j] = WMMA_BF16(a3, bf, acc[3][j]);
        }
    }

    if (o0 < 1024) {
        // ---- Q or K: destination rows are s, 64 contiguous d each ----------
#pragma unroll
        for (int ai = 0; ai < 4; ai++)
#pragma unroll
            for (int vv = 0; vv < 8; vv++) {
                int d_l = ai * 16 + vv + half * 8;
                float bi = bias[o0 + d_l];
#pragma unroll
                for (int j = 0; j < 4; j++)
                    T[(j * 16 + m) * 68 + d_l] = acc[ai][j][vv] + bi;   // T[s][d]
            }
        __bf16* dstbuf = (o0 < 512) ? qb : kb;
        const int hh = (o0 & 511) >> 6;
        const size_t rowbase = ((size_t)b * 8 + hh) * 1024 + s0;
#pragma unroll
        for (int rr = 0; rr < 2; rr++) {
            int s_l = rr * 32 + (l >> 1);
            int dh  = (l & 1) * 32;
            float buf[32];
#pragma unroll
            for (int j2 = 0; j2 < 8; j2++)
                *(float4*)&buf[j2 * 4] = *(const float4*)&T[s_l * 68 + dh + j2 * 4];
            __bf16* dst = dstbuf + (rowbase + s_l) * 64 + dh;
            *(uint4*)(dst + 0)  = pack8_bf16(buf + 0);
            *(uint4*)(dst + 8)  = pack8_bf16(buf + 8);
            *(uint4*)(dst + 16) = pack8_bf16(buf + 16);
            *(uint4*)(dst + 24) = pack8_bf16(buf + 24);
        }
    } else {
        // ---- V: destination rows are d, 64 contiguous t each ---------------
#pragma unroll
        for (int ai = 0; ai < 4; ai++)
#pragma unroll
            for (int vv = 0; vv < 8; vv++) {
                int d_l = ai * 16 + vv + half * 8;
                float bi = bias[o0 + d_l];
#pragma unroll
                for (int j = 0; j < 4; j++)
                    T[d_l * 68 + j * 16 + m] = acc[ai][j][vv] + bi;     // T[d][t]
            }
        const int hh = (o0 - 1024) >> 6;
        const size_t rowbase = ((size_t)b * 8 + hh) * 64;
#pragma unroll
        for (int rr = 0; rr < 2; rr++) {
            int d_l = rr * 32 + (l >> 1);
            int th  = (l & 1) * 32;
            float buf[32];
#pragma unroll
            for (int j2 = 0; j2 < 8; j2++)
                *(float4*)&buf[j2 * 4] = *(const float4*)&T[d_l * 68 + th + j2 * 4];
            __bf16* dst = vb + (rowbase + d_l) * 1024 + s0 + th;
            *(uint4*)(dst + 0)  = pack8_bf16(buf + 0);
            *(uint4*)(dst + 8)  = pack8_bf16(buf + 8);
            *(uint4*)(dst + 16) = pack8_bf16(buf + 16);
            *(uint4*)(dst + 24) = pack8_bf16(buf + 24);
        }
    }
}

// ---------------------------------------------------------------------------
// Kernel 3: attention. One wave per (b,h, 16-row s tile): grid = 64*64 = 4096.
// Logits (16 x 1024 f32, scaled) staged in 64KB LDS with XOR column swizzle
// (kills the bank conflicts of the 4KB row stride); two-lane-per-row softmax
// via wave32 __shfl_xor(..,16); PV with on-the-fly f32->bf16 A fragments.
// Output staged through LDS (reusing sc) for 64B-contiguous bf16 stores.
// ---------------------------------------------------------------------------
__global__ __launch_bounds__(32) void attn_kernel(const __bf16* __restrict__ qb,
                                                  const __bf16* __restrict__ kbuf,
                                                  const __bf16* __restrict__ vbuf,
                                                  __bf16* __restrict__ ao) {
    __shared__ float sc[16 * 1024];               // exactly 64 KB

    const int l = threadIdx.x;
    const int m = l & 15, half = l >> 4;
    int id = blockIdx.x;
    const int st = id & 63;
    const int bh = id >> 6;
    const int b = bh >> 3, h = bh & 7;
    const int s0 = st * 16;
    const float scale = 0.125f;                   // hd^-0.5, hd = 64

    // Q fragments for these 16 rows (K = d = 0..63 -> two 16x32 A frags)
    const __bf16* qrow = qb + ((size_t)bh * 1024 + s0) * 64;
    const v16bf aq0 = load_a_frag(qrow,      64, m, half * 16);
    const v16bf aq1 = load_a_frag(qrow + 32, 64, m, half * 16);

    // ---- logits = scale * Q^T K ------------------------------------------
    const __bf16* krow = kbuf + (size_t)bh * 1024 * 64;
    for (int t0 = 0; t0 < 1024; t0 += 16) {
        v8f acc = {};
        v16bf b0 = load_b_frag(krow + (size_t)t0 * 64,      64, m, half * 16);
        acc = WMMA_BF16(aq0, b0, acc);
        v16bf b1 = load_b_frag(krow + (size_t)t0 * 64 + 32, 64, m, half * 16);
        acc = WMMA_BF16(aq1, b1, acc);
#pragma unroll
        for (int vv = 0; vv < 8; vv++) {
            int row = vv + half * 8;
            sc[SW(row, t0 + m)] = acc[vv] * scale;
        }
    }

    // ---- softmax over t: lane pair (r, r+16) splits the 1024 columns -----
    const int r = m, cb = half * 512;
    float mx = -3.402823466e38f;
    for (int j = 0; j < 128; j++) {
        float4 f = *(const float4*)&sc[SW(r, cb + 4 * j)];
        mx = fmaxf(mx, fmaxf(fmaxf(f.x, f.y), fmaxf(f.z, f.w)));
    }
    mx = fmaxf(mx, __shfl_xor(mx, 16, 32));
    float sum = 0.f;
    for (int j = 0; j < 128; j++) {
        float4* p = (float4*)&sc[SW(r, cb + 4 * j)];
        float4 f = *p;
        f.x = __expf(f.x - mx); f.y = __expf(f.y - mx);
        f.z = __expf(f.z - mx); f.w = __expf(f.w - mx);
        sum += f.x + f.y + f.z + f.w;
        *p = f;
    }
    sum += __shfl_xor(sum, 16, 32);

    // ---- out = P @ V^T ----------------------------------------------------
    v8f accO[4] = {};
    const __bf16* vrow = vbuf + (size_t)bh * 64 * 1024;
    const int kg = half * 16;
    for (int t0 = 0; t0 < 1024; t0 += 32) {
        const float4* pp  = (const float4*)&sc[SW(m, t0 + kg)];
        const float4* pp2 = (const float4*)&sc[SW(m, t0 + kg + 8)];
        float4 f0 = pp[0], f1 = pp[1], f2 = pp2[0], f3 = pp2[1];
        const float tf[16] = { f0.x, f0.y, f0.z, f0.w, f1.x, f1.y, f1.z, f1.w,
                               f2.x, f2.y, f2.z, f2.w, f3.x, f3.y, f3.z, f3.w };
        v16bf ap;
#pragma unroll
        for (int i = 0; i < 16; i++) ap[i] = (__bf16)tf[i];
#pragma unroll
        for (int dt = 0; dt < 4; dt++) {
            v16bf bv = load_b_frag(vrow + (size_t)dt * 16 * 1024 + t0, 1024, m, half * 16);
            accO[dt] = WMMA_BF16(ap, bv, accO[dt]);
        }
    }

    // ---- normalize, transpose through LDS (reuse sc), wide stores --------
#pragma unroll
    for (int vv = 0; vv < 8; vv++) {
        const float rsum = __shfl(sum, vv + half * 8, 32);
        const float inv  = 1.0f / rsum;
        const int row    = vv + half * 8;
#pragma unroll
        for (int dt = 0; dt < 4; dt++)
            sc[row * 68 + dt * 16 + m] = accO[dt][vv] * inv;    // To[s_local][d]
    }
    {
        const int rr = l >> 1;            // s_local 0..15
        const int ch = (l & 1) * 32;      // d half
        float buf[32];
#pragma unroll
        for (int j2 = 0; j2 < 8; j2++)
            *(float4*)&buf[j2 * 4] = *(const float4*)&sc[rr * 68 + ch + j2 * 4];
        __bf16* dst = ao + ((size_t)b * 1024 + s0 + rr) * 512 + h * 64 + ch;
        *(uint4*)(dst + 0)  = pack8_bf16(buf + 0);
        *(uint4*)(dst + 8)  = pack8_bf16(buf + 8);
        *(uint4*)(dst + 16) = pack8_bf16(buf + 16);
        *(uint4*)(dst + 24) = pack8_bf16(buf + 24);
    }
}

// ---------------------------------------------------------------------------
// Kernel 4: proj GEMM + bias + residual, f32 output
// y[b,o,s] = x[b,o,s] + pb[o] + sum_c W[o,c] * ao[b,s,c]
// one wave per 64(o) x 64(s) tile; grid = 8 * 8 * 16 = 1024 blocks of 32.
// Epilogue transposes through LDS; residual/bias fused into coalesced writes.
// ---------------------------------------------------------------------------
__global__ __launch_bounds__(32) void proj_gemm_kernel(const __bf16* __restrict__ wp,
                                                       const __bf16* __restrict__ ao,
                                                       const float*  __restrict__ bias,
                                                       const float*  __restrict__ x,
                                                       float* __restrict__ out) {
    __shared__ float T[64 * 68];

    const int l = threadIdx.x;
    const int m = l & 15, half = l >> 4;
    const int kg = half * 16, kb16 = half * 16;
    int id = blockIdx.x;
    const int st4 = id & 15; id >>= 4;
    const int ot = id & 7;
    const int b  = id >> 3;
    const int o0 = ot * 64, s0 = st4 * 64;

    v8f acc[4][4] = {};
    const __bf16* wrow = wp + (size_t)o0 * 512;
    const __bf16* xb   = ao + ((size_t)b * 1024 + s0) * 512;

    for (int kc = 0; kc < 512; kc += 32) {
        __builtin_prefetch(wrow + (size_t)(m * 4) * 512 + kc + 64, 0, 1);
        __builtin_prefetch(xb + (size_t)(m * 4) * 512 + kc + 64, 0, 1);
        v16bf a0 = load_a_frag(wrow + kc,                    512, m, kg);
        v16bf a1 = load_a_frag(wrow + (size_t)16 * 512 + kc, 512, m, kg);
        v16bf a2 = load_a_frag(wrow + (size_t)32 * 512 + kc, 512, m, kg);
        v16bf a3 = load_a_frag(wrow + (size_t)48 * 512 + kc, 512, m, kg);
#pragma unroll
        for (int j = 0; j < 4; j++) {
            v16bf bf = load_b_frag(xb + (size_t)j * 16 * 512 + kc, 512, m, kb16);
            acc[0][j] = WMMA_BF16(a0, bf, acc[0][j]);
            acc[1][j] = WMMA_BF16(a1, bf, acc[1][j]);
            acc[2][j] = WMMA_BF16(a2, bf, acc[2][j]);
            acc[3][j] = WMMA_BF16(a3, bf, acc[3][j]);
        }
    }

    // stage T[o_local][s_local]
#pragma unroll
    for (int ai = 0; ai < 4; ai++)
#pragma unroll
        for (int vv = 0; vv < 8; vv++) {
            int o_l = ai * 16 + vv + half * 8;
#pragma unroll
            for (int j = 0; j < 4; j++)
                T[o_l * 68 + j * 16 + m] = acc[ai][j][vv];
        }

    // coalesced f32 writes: lane pair covers one o-row (256B contiguous)
#pragma unroll
    for (int rr = 0; rr < 2; rr++) {
        int o_l = rr * 32 + (l >> 1);
        int sh  = (l & 1) * 32;
        const float bi = bias[o0 + o_l];
        const size_t idx = ((size_t)b * 512 + o0 + o_l) * 1024 + s0 + sh;
#pragma unroll
        for (int j2 = 0; j2 < 8; j2++) {
            float4 t  = *(const float4*)&T[o_l * 68 + sh + j2 * 4];
            float4 xr = *(const float4*)&x[idx + j2 * 4];
            float4 o4;
            o4.x = t.x + xr.x + bi; o4.y = t.y + xr.y + bi;
            o4.z = t.z + xr.z + bi; o4.w = t.w + xr.w + bi;
            *(float4*)&out[idx + j2 * 4] = o4;
        }
    }
}

// ---------------------------------------------------------------------------
// Host-side launch
// ---------------------------------------------------------------------------
extern "C" void kernel_launch(void* const* d_in, const int* in_sizes, int n_in,
                              void* d_out, int out_size, void* d_ws, size_t ws_size,
                              hipStream_t stream) {
    (void)in_sizes; (void)n_in; (void)out_size; (void)ws_size;

    const float* x      = (const float*)d_in[0];
    const float* gamma  = (const float*)d_in[1];
    const float* beta   = (const float*)d_in[2];
    const float* qkv_w  = (const float*)d_in[3];
    const float* qkv_b  = (const float*)d_in[4];
    const float* proj_w = (const float*)d_in[5];
    const float* proj_b = (const float*)d_in[6];
    float* out          = (float*)d_out;

    // workspace layout (bf16 staging buffers)
    char* ws = (char*)d_ws;
    const size_t SZ_QKVW = (size_t)1536 * 512 * 2;     // 1.50 MB
    const size_t SZ_PRJW = (size_t)512  * 512 * 2;     // 0.50 MB
    const size_t SZ_ACT  = (size_t)8 * 1024 * 512 * 2; // 8 MB each
    __bf16* wq_bf = (__bf16*)(ws);
    __bf16* wp_bf = (__bf16*)(ws + SZ_QKVW);
    __bf16* xnt   = (__bf16*)(ws + SZ_QKVW + SZ_PRJW);
    __bf16* qb    = (__bf16*)(ws + SZ_QKVW + SZ_PRJW + 1 * SZ_ACT);
    __bf16* kb    = (__bf16*)(ws + SZ_QKVW + SZ_PRJW + 2 * SZ_ACT);
    __bf16* vb    = (__bf16*)(ws + SZ_QKVW + SZ_PRJW + 3 * SZ_ACT);
    __bf16* ao    = (__bf16*)(ws + SZ_QKVW + SZ_PRJW + 4 * SZ_ACT);

    // 0) weights -> bf16
    cvt_bf16_kernel<<<(1536 * 512 + 255) / 256, 256, 0, stream>>>(qkv_w, wq_bf, 1536 * 512);
    cvt_bf16_kernel<<<(512  * 512 + 255) / 256, 256, 0, stream>>>(proj_w, wp_bf, 512 * 512);

    // 1) GroupNorm -> transposed bf16 activations (b, s, c)
    gn_kernel<<<64, 256, 0, stream>>>(x, gamma, beta, xnt);

    // 2) QKV projection (WMMA, 64x64 tiles), scatter into attention layouts
    qkv_gemm_kernel<<<8 * 24 * 16, 32, 0, stream>>>(wq_bf, xnt, qkv_b, qb, kb, vb);

    // 3) attention (WMMA QK^T + swizzled-LDS softmax + WMMA PV)
    attn_kernel<<<64 * 64, 32, 0, stream>>>(qb, kb, vb, ao);

    // 4) output projection + bias + residual (WMMA, 64x64 tiles), f32 out
    proj_gemm_kernel<<<8 * 8 * 16, 32, 0, stream>>>(wp_bf, ao, proj_b, x, out);
}
